// DeepseekAttention_44899588112793
// MI455X (gfx1250) — matrix operand order repaired and checked by reading, not compiled, not run on hardware
//
#include <hip/hip_runtime.h>
#include <cstdint>

typedef __bf16 bf16_t;
typedef __attribute__((ext_vector_type(8)))  __bf16 v8bf;
typedef __attribute__((ext_vector_type(16))) __bf16 v16bf;
typedef __attribute__((ext_vector_type(8)))  float  v8f;
typedef __attribute__((ext_vector_type(4)))  unsigned int u32x4;
typedef __attribute__((ext_vector_type(8)))  int i32x8;
typedef __attribute__((ext_vector_type(4)))  int i32x4;

static __device__ __forceinline__ v16bf combine16(v8bf lo, v8bf hi) {
  v16bf r;
#pragma unroll
  for (int i = 0; i < 8; ++i) { r[i] = lo[i]; r[i + 8] = hi[i]; }
  return r;
}

// 16-lane butterfly max step via DPP (all lanes active).
template <int CTRL>
static __device__ __forceinline__ float fmax_dpp(float x) {
  int v = __builtin_amdgcn_mov_dpp(__builtin_bit_cast(int, x), CTRL, 0xF, 0xF, true);
  return fmaxf(x, __builtin_bit_cast(float, v));
}
static __device__ __forceinline__ float fmax_xor16(float x) {
  x = fmax_dpp<0xB1>(x);   // quad_perm(1,0,3,2)
  x = fmax_dpp<0x4E>(x);   // quad_perm(2,3,0,1)
  x = fmax_dpp<0x141>(x);  // row_half_mirror
  x = fmax_dpp<0x140>(x);  // row_mirror
  return x;
}

// ---------------------------------------------------------------------------
// TDM: DMA a 128-row x 32-col bf16 tile (row stride K elems) into LDS with
// 16B padding after every 64B row (-> 40-element LDS row stride).
// ---------------------------------------------------------------------------
static __device__ __forceinline__ void tdm_load_tile128x32(
    unsigned lds_off, const bf16_t* gbase, long row0, long k0, int K) {
  uint64_t gaddr = (uint64_t)(uintptr_t)gbase + (uint64_t)(row0 * (long)K + k0) * 2u;
  const unsigned rows = 1u << 20;
  u32x4 g0;
  g0[0] = 1u;                                        // count=1
  g0[1] = lds_off;                                   // lds_addr
  g0[2] = (unsigned)gaddr;                           // global_addr[31:0]
  g0[3] = (unsigned)(gaddr >> 32) | (2u << 30);      // global_addr[56:32] | type=2
  i32x8 g1;
  g1[0] = (int)((1u << 16) | (1u << 20) | (3u << 22) | (3u << 25));
  g1[1] = (int)(((unsigned)K & 0xFFFFu) << 16);      // tensor_dim0[15:0]
  g1[2] = (int)(((unsigned)K >> 16) | ((rows & 0xFFFFu) << 16));
  g1[3] = (int)((rows >> 16) | (32u << 16));         // tile_dim0=32
  g1[4] = 128;                                       // tile_dim1=128, tile_dim2=0
  g1[5] = K;                                         // tensor_dim0_stride[31:0]
  g1[6] = 0;
  g1[7] = 0;
  i32x4 z4 = {0, 0, 0, 0};
  i32x8 z8 = {0, 0, 0, 0, 0, 0, 0, 0};
  __builtin_amdgcn_tensor_load_to_lds(g0, g1, z4, z4, z8, 0);
}

// ---------------------------------------------------------------------------
// GEMM: C[M][N] = A[M][K] * W[N][K]^T (bf16 in via TDM double-buffer, WMMA,
// fp32 or bf16 out). Block 128x128, 8 waves (4M x 2N), wave tile 32x64.
// ---------------------------------------------------------------------------
template <bool BF16OUT>
__global__ __launch_bounds__(256) void gemm_tdm_wmma(
    const bf16_t* __restrict__ A, const bf16_t* __restrict__ W,
    void* __restrict__ Cv, int N, int K) {
  __shared__ __attribute__((aligned(16))) bf16_t As[2][128 * 40];
  __shared__ __attribute__((aligned(16))) bf16_t Ws[2][128 * 40];
  const int tid  = threadIdx.x;
  const int wave = tid >> 5, lane = tid & 31;
  const int hf   = lane >> 4, m16 = lane & 15;
  const int wm   = wave & 3,  wn  = wave >> 2;
  const long rowM = (long)blockIdx.y * 128;
  const long rowN = (long)blockIdx.x * 128;
  const int nk = K >> 5;
  v8f acc[2][4] = {};

  if (wave == 0) {
    tdm_load_tile128x32((unsigned)(uintptr_t)&As[0][0], A, rowM, 0, K);
    tdm_load_tile128x32((unsigned)(uintptr_t)&Ws[0][0], W, rowN, 0, K);
  }
  for (int it = 0; it < nk; ++it) {
    const int buf = it & 1;
    if (wave == 0) {
      if (it + 1 < nk) {
        tdm_load_tile128x32((unsigned)(uintptr_t)&As[buf ^ 1][0], A, rowM,
                            (long)(it + 1) << 5, K);
        tdm_load_tile128x32((unsigned)(uintptr_t)&Ws[buf ^ 1][0], W, rowN,
                            (long)(it + 1) << 5, K);
        __builtin_amdgcn_s_wait_tensorcnt(2);
      } else {
        __builtin_amdgcn_s_wait_tensorcnt(0);
      }
    }
    __syncthreads();  // tile `buf` visible to all waves
    v16bf af[2], bfm[4];
#pragma unroll
    for (int i = 0; i < 2; ++i) {
      const bf16_t* p = &As[buf][(wm * 32 + i * 16 + m16) * 40];
      af[i] = combine16(*(const v8bf*)&p[8 * hf], *(const v8bf*)&p[16 + 8 * hf]);
    }
#pragma unroll
    for (int j = 0; j < 4; ++j) {
      const bf16_t* p = &Ws[buf][(wn * 64 + j * 16 + m16) * 40];
      bfm[j] = combine16(*(const v8bf*)&p[8 * hf], *(const v8bf*)&p[16 + 8 * hf]);
    }
#pragma unroll
    for (int i = 0; i < 2; ++i)
#pragma unroll
      for (int j = 0; j < 4; ++j)
        acc[i][j] = __builtin_amdgcn_wmma_f32_16x16x32_bf16(
            false, af[i], false, bfm[j], (short)0, acc[i][j], false, false);
    __syncthreads();  // all reads of `buf` done before its TDM overwrite
  }
#pragma unroll
  for (int i = 0; i < 2; ++i)
#pragma unroll
    for (int j = 0; j < 4; ++j) {
      const int r0 = (int)rowM + wm * 32 + i * 16 + 8 * hf;
      const int c0 = (int)rowN + wn * 64 + j * 16 + m16;
      const int base = r0 * N + c0;
      if (BF16OUT) {
        bf16_t* cp = (bf16_t*)Cv;
#pragma unroll
        for (int r = 0; r < 8; ++r) cp[base + r * N] = (bf16_t)acc[i][j][r];
      } else {
        float* cp = (float*)Cv;
#pragma unroll
        for (int r = 0; r < 8; ++r) cp[base + r * N] = acc[i][j][r];
      }
    }
}

// fp32 -> bf16 elementwise (8 per thread, one b128 store)
__global__ void cvt_bf16_kernel(const float* __restrict__ in,
                                bf16_t* __restrict__ out, int n) {
  const int i = (blockIdx.x * 256 + threadIdx.x) * 8;
  if (i < n) {
    float4 a = *(const float4*)(in + i);
    float4 b = *(const float4*)(in + i + 4);
    v8bf o;
    o[0] = (bf16_t)a.x; o[1] = (bf16_t)a.y; o[2] = (bf16_t)a.z; o[3] = (bf16_t)a.w;
    o[4] = (bf16_t)b.x; o[5] = (bf16_t)b.y; o[6] = (bf16_t)b.z; o[7] = (bf16_t)b.w;
    *(v8bf*)(out + i) = o;
  }
}

// ---------------------------------------------------------------------------
// RoPE: q[s][32][128], k[s][8][128] bf16 -> bf16 head-major [h][S][128].
// Softmax scale 1/sqrt(128) is folded into Q here.
// ---------------------------------------------------------------------------
__global__ void rope_kernel(const bf16_t* __restrict__ qin,
                            const bf16_t* __restrict__ kin,
                            const int* __restrict__ pos_ids,
                            bf16_t* __restrict__ qout,
                            bf16_t* __restrict__ kout, int S) {
  const int s = blockIdx.x;
  const int d = threadIdx.x;  // 128 threads
  const float pos = (float)pos_ids[s];
  const int i = d & 63;
  const float inv = __expf(-(2.0f * (float)i / 128.0f) * 9.210340371976184f);
  const float ang = pos * inv;
  const float c = cosf(ang), sn = sinf(ang);
  const float scale = 0.08838834764831845f;  // 1/sqrt(128)
#pragma unroll 4
  for (int h = 0; h < 32; ++h) {
    const bf16_t* qp = qin + s * 4096 + h * 128;
    const float v = (float)qp[d];
    const float rot = (d < 64) ? -(float)qp[d + 64] : (float)qp[d - 64];
    qout[(h * S + s) * 128 + d] = (bf16_t)((v * c + rot * sn) * scale);
  }
#pragma unroll 4
  for (int h = 0; h < 8; ++h) {
    const bf16_t* kp = kin + s * 1024 + h * 128;
    const float v = (float)kp[d];
    const float rot = (d < 64) ? -(float)kp[d + 64] : (float)kp[d - 64];
    kout[(h * S + s) * 128 + d] = (bf16_t)(v * c + rot * sn);
  }
}

// v[s][8][128] bf16 -> vT[kvh][128][S] bf16
__global__ void vtrans_kernel(const bf16_t* __restrict__ vin,
                              bf16_t* __restrict__ vout, int S) {
  const int idx = blockIdx.x * 256 + threadIdx.x;  // S*1024 total
  const int d = idx & 127;
  const int t = idx >> 7;
  const int h = t & 7;
  const int s = t >> 3;
  vout[(h * 128 + d) * S + s] = vin[idx];
}

// ---------------------------------------------------------------------------
// One 32-key flash chunk: QK^T (8 WMMA) -> online softmax (DPP max, WMMA
// row-sum) -> P·V (8 WMMA). `masked` folds away causal compares except on
// the diagonal chunk.
// ---------------------------------------------------------------------------
static __device__ __forceinline__ void fa_chunk(
    int kb, bool masked, int qbase, int hf, int m16,
    const bf16_t* __restrict__ kbase, const bf16_t* __restrict__ vbase, int S,
    const v16bf (&qf)[4], const v16bf& ones, v8f (&out)[8], float (&mrow)[8],
    float (&lrow)[8], bf16_t (*pl)[40]) {
  v8f sc[2] = {};
#pragma unroll
  for (int t = 0; t < 2; ++t) {
    const bf16_t* kp = kbase + (kb + 16 * t + m16) * 128;
#pragma unroll
    for (int f = 0; f < 4; ++f) {
      v16bf kf = combine16(*(const v8bf*)&kp[32 * f + 8 * hf],
                           *(const v8bf*)&kp[32 * f + 16 + 8 * hf]);
      sc[t] = __builtin_amdgcn_wmma_f32_16x16x32_bf16(
          false, qf[f], false, kf, (short)0, sc[t], false, false);
    }
  }
  float corr[8];
#pragma unroll
  for (int r = 0; r < 8; ++r) {
    float s0 = sc[0][r];
    float s1 = sc[1][r];
    if (masked) {
      const int qrow = qbase + r + 8 * hf;
      if (kb + m16 > qrow)      s0 = -1e30f;
      if (kb + 16 + m16 > qrow) s1 = -1e30f;
    }
    const float tm = fmax_xor16(fmaxf(s0, s1));
    const float mnew = fmaxf(mrow[r], tm);
    const float c = __expf(mrow[r] - mnew);
    mrow[r] = mnew;
    corr[r] = c;
    const float p0 = __expf(s0 - mnew);
    const float p1 = __expf(s1 - mnew);
    const int prow = r + 8 * hf;
    pl[prow][m16]      = (bf16_t)p0;
    pl[prow][16 + m16] = (bf16_t)p1;
  }
#pragma unroll
  for (int t = 0; t < 8; ++t)
#pragma unroll
    for (int r = 0; r < 8; ++r) out[t][r] *= corr[r];
  asm volatile("s_wait_dscnt 0x0" ::: "memory");
  v16bf pf;
  {
    const bf16_t* pp = &pl[m16][0];
    pf = combine16(*(const v8bf*)&pp[8 * hf], *(const v8bf*)&pp[16 + 8 * hf]);
  }
  // Row sums of P via the matrix engine: P(16x32) x ones(32x16).
  v8f zero = {};
  v8f psum = __builtin_amdgcn_wmma_f32_16x16x32_bf16(
      false, pf, false, ones, (short)0, zero, false, false);
#pragma unroll
  for (int r = 0; r < 8; ++r) lrow[r] = lrow[r] * corr[r] + psum[r];
#pragma unroll
  for (int t = 0; t < 8; ++t) {
    const bf16_t* vp = vbase + (t * 16 + m16) * S + kb + 8 * hf;
    v16bf vf = combine16(*(const v8bf*)&vp[0], *(const v8bf*)&vp[16]);
    out[t] = __builtin_amdgcn_wmma_f32_16x16x32_bf16(
        false, pf, false, vf, (short)0, out[t], false, false);
  }
}

// ---------------------------------------------------------------------------
// Flash attention: wave owns 16 query rows, online softmax over 32-key chunks.
// Q[32][S][128] (pre-scaled), K[8][S][128], Vt[8][128][S] -> O[S][4096] bf16
// ---------------------------------------------------------------------------
__global__ __launch_bounds__(256) void flash_attn_wmma(
    const bf16_t* __restrict__ Q, const bf16_t* __restrict__ Kr,
    const bf16_t* __restrict__ Vt, bf16_t* __restrict__ O, int S) {
  __shared__ __attribute__((aligned(16))) bf16_t plds[8][16][40];
  const int h    = blockIdx.y;
  const int kvh  = h >> 2;  // n_rep = 4
  const int tid  = threadIdx.x;
  const int wave = tid >> 5, lane = tid & 31;
  const int hf   = lane >> 4, m16 = lane & 15;
  const int qbase = blockIdx.x * 128 + wave * 16;
  const bf16_t* kbase = Kr + kvh * S * 128;
  const bf16_t* vbase = Vt + kvh * 128 * S;

  v16bf qf[4];
  {
    const bf16_t* qp = Q + (h * S + qbase + m16) * 128;
#pragma unroll
    for (int f = 0; f < 4; ++f)
      qf[f] = combine16(*(const v8bf*)&qp[32 * f + 8 * hf],
                        *(const v8bf*)&qp[32 * f + 16 + 8 * hf]);
  }
  v16bf ones;
#pragma unroll
  for (int i = 0; i < 16; ++i) ones[i] = (bf16_t)1.0f;

  float mrow[8], lrow[8];
  v8f out[8] = {};
#pragma unroll
  for (int r = 0; r < 8; ++r) { mrow[r] = -1e30f; lrow[r] = 0.0f; }

  const int kb_diag = qbase & ~31;  // only chunk that needs causal masking
  for (int kb = 0; kb < kb_diag; kb += 32)
    fa_chunk(kb, false, qbase, hf, m16, kbase, vbase, S, qf, ones, out, mrow,
             lrow, plds[wave]);
  fa_chunk(kb_diag, true, qbase, hf, m16, kbase, vbase, S, qf, ones, out, mrow,
           lrow, plds[wave]);

#pragma unroll
  for (int r = 0; r < 8; ++r) {
    const float inv = __builtin_amdgcn_rcpf(lrow[r]);
    const int srow = qbase + r + 8 * hf;
#pragma unroll
    for (int t = 0; t < 8; ++t)
      O[srow * 4096 + h * 128 + t * 16 + m16] = (bf16_t)(out[t][r] * inv);
  }
}

// ---------------------------------------------------------------------------
extern "C" void kernel_launch(void* const* d_in, const int* in_sizes, int n_in,
                              void* d_out, int out_size, void* d_ws, size_t ws_size,
                              hipStream_t stream) {
  const float* hidden  = (const float*)d_in[0];
  // d_in[1] = attention_mask (causal; applied analytically in-kernel)
  const int*   pos_ids = (const int*)d_in[2];
  const float* Wq      = (const float*)d_in[3];
  const float* Wk      = (const float*)d_in[4];
  const float* Wv      = (const float*)d_in[5];
  const float* Wo      = (const float*)d_in[6];
  float* out = (float*)d_out;

  const int S = 2048, H = 4096, KVH = 1024;
  char* ws = (char*)d_ws;
  bf16_t* hid_bf = (bf16_t*)(ws);                    // 16 MB [S][4096]
  bf16_t* Wq_bf  = (bf16_t*)(ws + 16777216);         // 32 MB
  bf16_t* Wk_bf  = (bf16_t*)(ws + 50331648);         //  8 MB
  bf16_t* Wv_bf  = (bf16_t*)(ws + 58720256);         //  8 MB
  bf16_t* Wo_bf  = (bf16_t*)(ws + 67108864);         // 32 MB
  bf16_t* q_bf   = (bf16_t*)(ws + 100663296);        // 16 MB [S][4096]
  bf16_t* k_bf   = (bf16_t*)(ws + 117440512);        //  4 MB [S][1024]
  bf16_t* v_bf   = (bf16_t*)(ws + 121634816);        //  4 MB [S][1024]
  bf16_t* q_rope = (bf16_t*)(ws + 125829120);        // 16 MB [32][S][128]
  bf16_t* k_rope = (bf16_t*)(ws + 142606336);        //  4 MB [8][S][128]
  bf16_t* vT     = (bf16_t*)(ws + 146800640);        //  4 MB [8][128][S]
  bf16_t* attn   = (bf16_t*)(ws);                    // 16 MB, aliases hid_bf (dead by then)

  dim3 blk(256);
  // fp32 -> bf16 operand prep
  cvt_bf16_kernel<<<dim3((S * H)   / 2048), blk, 0, stream>>>(hidden, hid_bf, S * H);
  cvt_bf16_kernel<<<dim3((H * H)   / 2048), blk, 0, stream>>>(Wq, Wq_bf, H * H);
  cvt_bf16_kernel<<<dim3((KVH * H) / 2048), blk, 0, stream>>>(Wk, Wk_bf, KVH * H);
  cvt_bf16_kernel<<<dim3((KVH * H) / 2048), blk, 0, stream>>>(Wv, Wv_bf, KVH * H);
  cvt_bf16_kernel<<<dim3((H * H)   / 2048), blk, 0, stream>>>(Wo, Wo_bf, H * H);
  // Projections (TDM-staged WMMA GEMMs, bf16 out)
  gemm_tdm_wmma<true><<<dim3(H / 128,   S / 128), blk, 0, stream>>>(hid_bf, Wq_bf, q_bf, H,   H);
  gemm_tdm_wmma<true><<<dim3(KVH / 128, S / 128), blk, 0, stream>>>(hid_bf, Wk_bf, k_bf, KVH, H);
  gemm_tdm_wmma<true><<<dim3(KVH / 128, S / 128), blk, 0, stream>>>(hid_bf, Wv_bf, v_bf, KVH, H);
  // RoPE + layout transforms
  rope_kernel<<<dim3(S), dim3(128), 0, stream>>>(q_bf, k_bf, pos_ids, q_rope, k_rope, S);
  vtrans_kernel<<<dim3((S * KVH) / 256), blk, 0, stream>>>(v_bf, vT, S);
  // Fused causal attention (bf16 out)
  flash_attn_wmma<<<dim3(S / 128, 32), blk, 0, stream>>>(q_rope, k_rope, vT, attn, S);
  // Output projection (fp32 out)
  gemm_tdm_wmma<false><<<dim3(H / 128, S / 128), blk, 0, stream>>>(attn, Wo_bf, out, H, H);
}
